// HeteroGNN_62079457296455
// MI455X (gfx1250) — matrix operand order repaired and checked by reading, not compiled, not run on hardware
//
#include <hip/hip_runtime.h>
#include <hip/hip_bf16.h>

typedef __attribute__((ext_vector_type(16))) __bf16 v16bf;
typedef __attribute__((ext_vector_type(8)))  __bf16 bf16x8;
typedef __attribute__((ext_vector_type(8)))  float  v8f;

#define NODES 50000
#define DIM   128
#define HID   256
#define OUTD  128
#define NEDGE 500000

// ---------------------------------------------------------------------------
// Zero fp32 buffer (float4 per thread)
// ---------------------------------------------------------------------------
__global__ void zero_f32(float* __restrict__ p, int n) {
    int i = (blockIdx.x * blockDim.x + threadIdx.x) * 4;
    if (i + 3 < n) {
        *(float4*)(p + i) = make_float4(0.f, 0.f, 0.f, 0.f);
    } else {
        for (int j = i; j < n; ++j) p[j] = 0.f;
    }
}

// ---------------------------------------------------------------------------
// Edge scatter: 32 lanes per edge, 4 floats per lane; fp32 atomic accumulate
// ---------------------------------------------------------------------------
__global__ void edge_scatter(const float* __restrict__ x_src,
                             const int*   __restrict__ ei,
                             const float* __restrict__ w,
                             float* __restrict__ agg, int E) {
    int idx  = blockIdx.x * blockDim.x + threadIdx.x;
    int e    = idx >> 5;
    int f    = (idx & 31) * 4;
    if (e >= E) return;
    int   s  = ei[e];
    int   d  = ei[E + e];
    float wt = w[e];
    float4 v = *(const float4*)(x_src + (size_t)s * DIM + f);
    float* a = agg + (size_t)d * DIM + f;
    atomicAdd(a + 0, wt * v.x);
    atomicAdd(a + 1, wt * v.y);
    atomicAdd(a + 2, wt * v.z);
    atomicAdd(a + 3, wt * v.w);
}

// ---------------------------------------------------------------------------
// fp32 -> bf16 conversion (4 per thread)
// ---------------------------------------------------------------------------
__global__ void cvt_f32_bf16(const float* __restrict__ in, __bf16* __restrict__ out, int n) {
    int i = (blockIdx.x * blockDim.x + threadIdx.x) * 4;
    if (i + 3 < n) {
        float4 v = *(const float4*)(in + i);
        out[i + 0] = (__bf16)v.x;
        out[i + 1] = (__bf16)v.y;
        out[i + 2] = (__bf16)v.z;
        out[i + 3] = (__bf16)v.w;
    } else {
        for (int j = i; j < n; ++j) out[j] = (__bf16)in[j];
    }
}

// ---------------------------------------------------------------------------
// Build fused B matrix [HID x 384] = [Wr_a | Wr_b | Ws_a + Ws_b] in bf16,
// plus fused bias = br_a + br_b (fp32). Pointers pre-offset to layer L-1.
// ---------------------------------------------------------------------------
__global__ void build_B(const float* __restrict__ Wr_a, const float* __restrict__ Wr_b,
                        const float* __restrict__ Ws_a, const float* __restrict__ Ws_b,
                        const float* __restrict__ br_a, const float* __restrict__ br_b,
                        __bf16* __restrict__ B, float* __restrict__ bias) {
    int idx = blockIdx.x * blockDim.x + threadIdx.x;
    if (idx >= HID * 384) return;
    int h = idx / 384;
    int k = idx % 384;
    float v;
    if      (k < 128) v = Wr_a[h * DIM + k];
    else if (k < 256) v = Wr_b[h * DIM + (k - 128)];
    else              v = Ws_a[h * DIM + (k - 256)] + Ws_b[h * DIM + (k - 256)];
    B[idx] = (__bf16)v;
    if (k == 0) bias[h] = br_a[h] + br_b[h];
}

// ---------------------------------------------------------------------------
// LDS byte offset of a generic pointer into __shared__: the LDS aperture
// lives entirely in addr[63:32] (ISA §10.2), so addr[31:0] IS the DS address.
// ---------------------------------------------------------------------------
__device__ __forceinline__ unsigned lds_off_u32(const void* p) {
    return (unsigned)(unsigned long long)p;
}

// ---------------------------------------------------------------------------
// WMMA GEMM (persistent blocks, full-B staged in LDS via CDNA5 async DMA):
//   out[M x NOUT] = relu( A[M x K] * W[NOUT x K]^T + bias )
// A is split into up to 3 segments of 128 K-columns each (ptr + row stride).
// A whole workgroup (8 waves) stages B [NOUT x K] bf16 into LDS once with
// global_load_async_to_lds_b128 (ASYNCcnt, no VGPR round-trip; CDNA5 320KB
// LDS per workgroup: 256x392x2 = 200KB fits), then grid-strides over 16-row
// slabs. Per slab: A fragments for the full K depth are register resident;
// B fragments stream from LDS via ds_load_b128, software-pipelined one step
// ahead of the WMMA so the DS fetch hides under the matrix op. The +8-element
// row pitch makes reads conflict-free ((K+8)*2/4 mod 64 == 4 -> 16 lanes
// tile all 64 banks).
// Fragment layouts per CDNA5 ISA 7.12.2:
//   A (16x32 bf16): lane = m%16 (+hi half), elem j -> K = (j%8) + 16*(j/8) + 8*hi
//   B (32x16 bf16): lane = n%16 (+hi half), elem j -> K = (j%8) + 16*(j/8) + 8*hi
//   C/D (16x16 f32): lane = n (+hi), vgpr v -> M = v + 8*hi
// ---------------------------------------------------------------------------
template <int K, int NOUT, bool OUT_BF16>
__global__ void gemm_bias_relu(const __bf16* __restrict__ a0, int lda0,
                               const __bf16* __restrict__ a1, int lda1,
                               const __bf16* __restrict__ a2, int lda2,
                               const __bf16* __restrict__ Bm,
                               const float*  __restrict__ bias,
                               void* __restrict__ outv, int M) {
    constexpr int KSTEPS = K / 32;
    constexpr int PITCH  = K + 8;               // LDS row pitch (bank-conflict-free)
    extern __shared__ __bf16 Bs[];              // NOUT * PITCH bf16

    // ---- async staging of the whole B matrix into LDS (ASYNCcnt path) ----
    // NOUT*CPR is an exact multiple of blockDim.x, so EXEC stays all-ones.
    constexpr int CPR = K / 8;                  // 16B chunks per row
    for (int c = threadIdx.x; c < NOUT * CPR; c += blockDim.x) {
        const int row = c / CPR;
        const int ci  = c % CPR;
        const __bf16* g = Bm + (size_t)row * K + ci * 8;
        unsigned lds = lds_off_u32(&Bs[row * PITCH + ci * 8]);
        asm volatile("global_load_async_to_lds_b128 %0, %1, off"
                     :: "v"(lds), "v"(g) : "memory");
    }
    asm volatile("s_wait_asynccnt 0x0" ::: "memory");
    __syncthreads();

    const int lane   = threadIdx.x & 31;
    const int wave   = threadIdx.x >> 5;
    const int nwaves = blockDim.x >> 5;
    const int ln     = lane & 15;
    const int hi     = lane >> 4;
    const int rowTiles   = M / 16;
    const int tileStride = gridDim.x * nwaves;

    const __bf16* segp[3] = {a0, a1, a2};
    const int     segs[3] = {lda0, lda1, lda2};

    for (int rt = blockIdx.x * nwaves + wave; rt < rowTiles; rt += tileStride) {
        const int m0   = rt * 16;
        const int mrow = m0 + ln;

        // Load all A fragments (register resident for entire NOUT sweep)
        v16bf afrag[KSTEPS];
#pragma unroll
        for (int s = 0; s < KSTEPS; ++s) {
            const int kb  = s * 32;
            const int seg = kb >> 7;
            const __bf16* base = segp[seg] + (size_t)mrow * segs[seg] + (kb & 127) + 8 * hi;
            bf16x8 lo  = *(const bf16x8*)(base);
            bf16x8 hi8 = *(const bf16x8*)(base + 16);
#pragma unroll
            for (int j = 0; j < 8; ++j) { afrag[s][j] = lo[j]; afrag[s][8 + j] = hi8[j]; }
        }

        // Prefetch next persistent tile's A rows into cache while we compute
        const int nextm = mrow + tileStride * 16;
        if (nextm < M) {
            __builtin_prefetch(segp[0] + (size_t)nextm * segs[0], 0, 1);
            __builtin_prefetch(segp[1] + (size_t)nextm * segs[1], 0, 1);
            __builtin_prefetch(segp[2] + (size_t)nextm * segs[2], 0, 1);
        }

        for (int nt = 0; nt < NOUT / 16; ++nt) {
            const int n = nt * 16 + ln;
            const float bv = bias[n];
            v8f acc = {bv, bv, bv, bv, bv, bv, bv, bv};

            const __bf16* brow = &Bs[n * PITCH + 8 * hi];
            // Software pipeline: B fragment for step s+1 is in flight while
            // the WMMA for step s executes (partial DScnt wait).
            bf16x8 lo = *(const bf16x8*)(brow);
            bf16x8 h8 = *(const bf16x8*)(brow + 16);
#pragma unroll
            for (int s = 0; s < KSTEPS; ++s) {
                v16bf bfrag;
#pragma unroll
                for (int j = 0; j < 8; ++j) { bfrag[j] = lo[j]; bfrag[8 + j] = h8[j]; }
                bf16x8 nlo = {}, nh8 = {};
                if (s + 1 < KSTEPS) {
                    const __bf16* bb = brow + (s + 1) * 32;
                    nlo = *(const bf16x8*)(bb);
                    nh8 = *(const bf16x8*)(bb + 16);
                }
                acc = __builtin_amdgcn_wmma_f32_16x16x32_bf16(
                    false, afrag[s], false, bfrag, (short)0, acc, false, false);
                lo = nlo; h8 = nh8;
            }

#pragma unroll
            for (int v = 0; v < 8; ++v) {
                float r = acc[v] > 0.f ? acc[v] : 0.f;
                const int m = m0 + v + 8 * hi;
                const size_t off = (size_t)m * NOUT + n;
                if constexpr (OUT_BF16) ((__bf16*)outv)[off] = (__bf16)r;
                else                    ((float*)outv)[off]  = r;
            }
        }
    }
}

// ---------------------------------------------------------------------------
extern "C" void kernel_launch(void* const* d_in, const int* in_sizes, int n_in,
                              void* d_out, int out_size, void* d_ws, size_t ws_size,
                              hipStream_t stream) {
    // Input order: x_gen, x_rain, then per relation r in {gg,gr,rg,rr}:
    //   ei_r, w_r, Wr_r, br_r, Ws_r; then W_gen, b_gen, W_rain, b_rain.
    const float* x_gen  = (const float*)d_in[0];
    const float* x_rain = (const float*)d_in[1];
    const int*   ei_gg = (const int*)d_in[2];  const float* w_gg = (const float*)d_in[3];
    const float* Wr_gg = (const float*)d_in[4]; const float* br_gg = (const float*)d_in[5];
    const float* Ws_gg = (const float*)d_in[6];
    const int*   ei_gr = (const int*)d_in[7];  const float* w_gr = (const float*)d_in[8];
    const float* Wr_gr = (const float*)d_in[9]; const float* br_gr = (const float*)d_in[10];
    const float* Ws_gr = (const float*)d_in[11];
    const int*   ei_rg = (const int*)d_in[12]; const float* w_rg = (const float*)d_in[13];
    const float* Wr_rg = (const float*)d_in[14]; const float* br_rg = (const float*)d_in[15];
    const float* Ws_rg = (const float*)d_in[16];
    const int*   ei_rr = (const int*)d_in[17]; const float* w_rr = (const float*)d_in[18];
    const float* Wr_rr = (const float*)d_in[19]; const float* br_rr = (const float*)d_in[20];
    const float* Ws_rr = (const float*)d_in[21];
    const float* W_gen  = (const float*)d_in[22]; const float* b_gen  = (const float*)d_in[23];
    const float* W_rain = (const float*)d_in[24]; const float* b_rain = (const float*)d_in[25];

    const size_t LOFF  = (size_t)HID * DIM;  // layer-1 offset into (L,H,D)
    const size_t LBOFF = HID;                // layer-1 offset into (L,H)

    // Workspace layout
    char* ws = (char*)d_ws;
    const size_t NF = (size_t)NODES * DIM;            // 6.4M
    float*  agg_f   = (float*)ws;                     // 4 * NF fp32 (102.4 MB)
    float*  agg_gg_f = agg_f;
    float*  agg_rg_f = agg_f + NF;
    float*  agg_gr_f = agg_f + 2 * NF;
    float*  agg_rr_f = agg_f + 3 * NF;
    // h buffers reuse the (dead after conversion) fp32 agg region
    __bf16* h_gen  = (__bf16*)ws;
    __bf16* h_rain = (__bf16*)(ws + (size_t)NODES * HID * 2);
    size_t o1 = 4 * NF * 4;
    __bf16* agg_bf    = (__bf16*)(ws + o1);           // 4 * NF bf16
    __bf16* agg_gg_bf = agg_bf;
    __bf16* agg_rg_bf = agg_bf + NF;
    __bf16* agg_gr_bf = agg_bf + 2 * NF;
    __bf16* agg_rr_bf = agg_bf + 3 * NF;
    size_t o2 = o1 + 4 * NF * 2;
    __bf16* x_gen_bf  = (__bf16*)(ws + o2);
    __bf16* x_rain_bf = x_gen_bf + NF;
    size_t o3 = o2 + 2 * NF * 2;
    __bf16* B_gen  = (__bf16*)(ws + o3);              // HID x 384
    __bf16* B_rain = B_gen + (size_t)HID * 384;
    __bf16* Wg_bf  = B_rain + (size_t)HID * 384;      // OUTD x HID
    __bf16* Wrn_bf = Wg_bf + (size_t)OUTD * HID;
    float*  bias_gen  = (float*)(Wrn_bf + (size_t)OUTD * HID);
    float*  bias_rain = bias_gen + HID;

    // 1) zero fp32 aggregation buffers (must happen every call)
    {
        int n = (int)(4 * NF);
        zero_f32<<<(n / 4 + 255) / 256, 256, 0, stream>>>(agg_f, n);
    }

    // 2) edge scatter per relation (dst gets w * x_src[src])
    {
        int thr = NEDGE * 32;
        int blk = (thr + 255) / 256;
        edge_scatter<<<blk, 256, 0, stream>>>(x_gen,  ei_gg, w_gg, agg_gg_f, NEDGE); // gen->gen
        edge_scatter<<<blk, 256, 0, stream>>>(x_rain, ei_rg, w_rg, agg_rg_f, NEDGE); // rain->gen
        edge_scatter<<<blk, 256, 0, stream>>>(x_gen,  ei_gr, w_gr, agg_gr_f, NEDGE); // gen->rain
        edge_scatter<<<blk, 256, 0, stream>>>(x_rain, ei_rr, w_rr, agg_rr_f, NEDGE); // rain->rain
    }

    // 3) conversions to bf16
    {
        int n = (int)(4 * NF);
        cvt_f32_bf16<<<(n / 4 + 255) / 256, 256, 0, stream>>>(agg_f, agg_bf, n);
        cvt_f32_bf16<<<((int)NF / 4 + 255) / 256, 256, 0, stream>>>(x_gen,  x_gen_bf,  (int)NF);
        cvt_f32_bf16<<<((int)NF / 4 + 255) / 256, 256, 0, stream>>>(x_rain, x_rain_bf, (int)NF);
        int nw = OUTD * HID;
        cvt_f32_bf16<<<(nw / 4 + 255) / 256, 256, 0, stream>>>(W_gen,  Wg_bf,  nw);
        cvt_f32_bf16<<<(nw / 4 + 255) / 256, 256, 0, stream>>>(W_rain, Wrn_bf, nw);
    }

    // 4) fused B matrices + biases (layer L-1)
    {
        int n = HID * 384;
        build_B<<<(n + 255) / 256, 256, 0, stream>>>(
            Wr_gg + LOFF, Wr_rg + LOFF, Ws_gg + LOFF, Ws_rg + LOFF,
            br_gg + LBOFF, br_rg + LBOFF, B_gen, bias_gen);
        build_B<<<(n + 255) / 256, 256, 0, stream>>>(
            Wr_gr + LOFF, Wr_rr + LOFF, Ws_gr + LOFF, Ws_rr + LOFF,
            br_gr + LBOFF, br_rr + LBOFF, B_rain, bias_rain);
    }

    // 5) hidden-layer GEMMs: h = relu([agg_a|agg_b|x] @ B^T + bias), bf16 out
    //    Full B (256x392 bf16 = 200KB) staged in LDS per workgroup (CDNA5 320KB).
    {
        constexpr int SH1 = HID * (384 + 8) * 2;   // 200704 B
        (void)hipFuncSetAttribute(
            reinterpret_cast<const void*>(&gemm_bias_relu<384, HID, true>),
            hipFuncAttributeMaxDynamicSharedMemorySize, SH1);
        gemm_bias_relu<384, HID, true><<<256, 256, SH1, stream>>>(
            agg_gg_bf, DIM, agg_rg_bf, DIM, x_gen_bf, DIM,
            B_gen, bias_gen, (void*)h_gen, NODES);
        gemm_bias_relu<384, HID, true><<<256, 256, SH1, stream>>>(
            agg_gr_bf, DIM, agg_rr_bf, DIM, x_rain_bf, DIM,
            B_rain, bias_rain, (void*)h_rain, NODES);
    }

    // 6) output GEMMs: out = relu(h @ W^T + b), fp32 out into d_out
    {
        float* out_gen  = (float*)d_out;
        float* out_rain = out_gen + (size_t)NODES * OUTD;
        constexpr int SH2 = OUTD * (256 + 8) * 2;  // 67584 B
        (void)hipFuncSetAttribute(
            reinterpret_cast<const void*>(&gemm_bias_relu<256, OUTD, false>),
            hipFuncAttributeMaxDynamicSharedMemorySize, SH2);
        gemm_bias_relu<256, OUTD, false><<<256, 256, SH2, stream>>>(
            h_gen, HID, h_gen + 128, HID, h_gen, HID,
            Wg_bf, b_gen, (void*)out_gen, NODES);
        gemm_bias_relu<256, OUTD, false><<<256, 256, SH2, stream>>>(
            h_rain, HID, h_rain + 128, HID, h_rain, HID,
            Wrn_bf, b_rain, (void*)out_rain, NODES);
    }
}
